// OccupancyDecoder2D_38371237823020
// MI455X (gfx1250) — matrix operand order repaired and checked by reading, not compiled, not run on hardware
//
#include <hip/hip_runtime.h>
#include <hip/hip_bf16.h>
#include <math.h>

typedef __attribute__((ext_vector_type(16))) _Float16 v16h;
typedef __attribute__((ext_vector_type(8)))  _Float16 v8h;
typedef __attribute__((ext_vector_type(2)))  __fp16   v2fp;   // cvt_pkrtz return type
typedef __attribute__((ext_vector_type(8)))  float    v8f;

#define PI_F 3.14159265358979323846f

// Weights are stored as A-matrix fragments of W^T (16 neurons x K), f16:
//   A-frag (16x32) element (lane L, half i):
//     m_local = L%16 (neuron-in-tile), g = L/16
//     k = 32c + 16g + i
// Layer1: K=64 (padded; k=56 row holds b1, "1" input), 8 tiles x 2 chunks
// Layer2: K=128, 4 tiles x 4 chunks ; Layer3: K=64, 2 tiles x 2 chunks
#define F1_OFF 0
#define F2_OFF 8192
#define F3_OFF 16384
#define FRAG_TOTAL 18432

// ---------------- prep: swizzle W^T (+b1) into f16 A-fragments ----------------
__global__ void occdec_prep(const float* __restrict__ W1,
                            const float* __restrict__ b1,
                            const float* __restrict__ W2,
                            const float* __restrict__ W3,
                            _Float16* __restrict__ frags) {
  int tt = blockIdx.x * blockDim.x + threadIdx.x;
  if (tt >= FRAG_TOTAL) return;

  int layer, t, c, local;
  if (tt < F2_OFF)      { int f = tt / 512; local = tt % 512; layer = 1; t = f >> 1; c = f & 1; }
  else if (tt < F3_OFF) { int u = tt - F2_OFF; int f = u / 512; local = u % 512; layer = 2; t = f >> 2; c = f & 3; }
  else                  { int u = tt - F3_OFF; int f = u / 512; local = u % 512; layer = 3; t = f >> 1; c = f & 1; }

  int L = local >> 4;                   // lane 0..31
  int i = local & 15;                   // half 0..15
  int g = L >> 4;
  int k = 32 * c + 16 * g + i;
  int neuron = 16 * t + (L & 15);

  float v;
  if (layer == 1) {
    if (k < 56)       v = W1[(size_t)k * 128 + neuron];
    else if (k == 56) v = b1[neuron];            // bias row (input col 56 == 1)
    else              v = 0.0f;
  } else if (layer == 2) {
    v = W2[(size_t)k * 64 + neuron];
  } else {
    v = W3[(size_t)k * 32 + neuron];
  }
  frags[tt] = (_Float16)v;
}

// ---------------- helpers ----------------
union U16 { v16h h; unsigned u[8]; };
union U8  { v8h  h; unsigned u[4]; };
union U2  { v2fp h; unsigned u;    };
union UF  { float f; unsigned u;   };

__device__ inline v8f wmma_f16(v16h a, v16h b, v8f c) {
  return __builtin_amdgcn_wmma_f32_16x16x32_f16(false, a, false, b,
                                                (short)0, c, false, false);
}

// pure-VALU cross-half swap (lane <-> lane^16): v_permlanex16_b32, identity sel
__device__ inline unsigned xor16(unsigned x) {
  return __builtin_amdgcn_permlanex16(x, x, 0x76543210u, 0xfedcba98u,
                                      false, false);
}
__device__ inline float xor16f(float x) {
  UF a; a.f = x; a.u = xor16(a.u); return a.f;
}

// pack v8f -> v8h with v_cvt_pk_rtz_f16_f32, then ReLU as packed f16 max
__device__ inline v8h relu_pack(v8f a) {
  U8 r;
#pragma unroll
  for (int w = 0; w < 4; ++w) {
    U2 p;
    p.h = __builtin_amdgcn_cvt_pkrtz(a[2 * w], a[2 * w + 1]);
    r.u[w] = p.u;
  }
  v8h z = {};
  return __builtin_elementwise_max(r.h, z);   // v_pk_max_num_f16 x4
}

// Build next-layer B-fragment (k-chunk of 32) from two C tiles (2c, 2c+1)
// packed as v8h per lane. Cross-lane exchange is only with lane^16.
__device__ inline v16h shuffle_frag(v8h pa, v8h pb, int g) {
  U8 ua, ub; ua.h = pa; ub.h = pb;
  U16 f;
#pragma unroll
  for (int w = 0; w < 4; ++w) {
    unsigned x0 = xor16(ua.u[w]);
    unsigned x1 = xor16(ub.u[w]);
    f.u[w]     = g ? x1      : ua.u[w];
    f.u[4 + w] = g ? ub.u[w] : x0;
  }
  return f.h;
}

__device__ inline void dbl_angle(float& s, float& c) {
  float s2 = 2.0f * s * c;
  float c2 = 1.0f - 2.0f * s * s;
  s = s2; c = c2;
}

// ---------------- main fused MLP kernel ----------------
__global__ void __launch_bounds__(256)
__attribute__((amdgpu_waves_per_eu(8)))
occdec_main(const float* __restrict__ gf,   // [M,16]
            const float* __restrict__ qp,   // [Q,2]
            const float* __restrict__ b2,
            const float* __restrict__ b3,
            const float* __restrict__ W4,   // [32]
            const float* __restrict__ b4,   // [1]
            const _Float16* __restrict__ frags,
            float* __restrict__ out,
            int Q, int M, int nTiles) {
  __shared__ __align__(32) _Float16 sW[FRAG_TOTAL];  // 36 KB weight A-frags

  const int tid = threadIdx.x;
  {  // vectorized weight staging: 36864 B = 2304 uint4
    const uint4* src = (const uint4*)frags;
    uint4* dst = (uint4*)sW;
    for (int t = tid; t < 2304; t += 256) dst[t] = src[t];
  }
  __syncthreads();

  const int lane = tid & 31;
  const int wave = tid >> 5;
  const int g    = lane >> 4;   // lane-pair group
  const int lm   = lane & 15;   // query column slot

  // per-lane loop-invariant bias / W4 vectors: element r maps to neuron 16t+r+8g
  v8f bias2v[4], bias3v[2];
  float w4v[2][8];
#pragma unroll
  for (int t = 0; t < 4; ++t) {
    const float4* p = (const float4*)(b2 + 16 * t + 8 * g);
    float4 a = p[0], b = p[1];
    bias2v[t][0] = a.x; bias2v[t][1] = a.y; bias2v[t][2] = a.z; bias2v[t][3] = a.w;
    bias2v[t][4] = b.x; bias2v[t][5] = b.y; bias2v[t][6] = b.z; bias2v[t][7] = b.w;
  }
#pragma unroll
  for (int t = 0; t < 2; ++t) {
    const float4* p = (const float4*)(b3 + 16 * t + 8 * g);
    float4 a = p[0], b = p[1];
    bias3v[t][0] = a.x; bias3v[t][1] = a.y; bias3v[t][2] = a.z; bias3v[t][3] = a.w;
    bias3v[t][4] = b.x; bias3v[t][5] = b.y; bias3v[t][6] = b.z; bias3v[t][7] = b.w;
    const float4* w = (const float4*)(W4 + 16 * t + 8 * g);
    float4 wa = w[0], wb = w[1];
    w4v[t][0] = wa.x; w4v[t][1] = wa.y; w4v[t][2] = wa.z; w4v[t][3] = wa.w;
    w4v[t][4] = wb.x; w4v[t][5] = wb.y; w4v[t][6] = wb.z; w4v[t][7] = wb.w;
  }
  const float b4v = b4[0];

  const int waveGlobal = blockIdx.x * 8 + wave;
  const int waveStride = gridDim.x * 8;

  for (int tile = waveGlobal; tile < nTiles; tile += waveStride) {
    const int q0 = tile * 16;
    const int q  = q0 + lm;                 // this lane's query
    const int qc = (q < Q) ? q : (Q - 1);   // clamp for ragged tail

    // ---- build B = X^T fragments in registers (no LDS) ----
    // B-frag chunk c, lane (lm,g), half i: k = 32c + 16g + i, query = q0 + lm.
    // k: 0..15 feats | 16..55 enc | 56 = 1 (bias input) | 57..63 = 0
    v16h Bx0, Bx1;
    {
      const float p0 = qp[(size_t)qc * 2 + 0];
      const float p1 = qp[(size_t)qc * 2 + 1];
      if (g == 0) {
        // chunk0: features 0..15
        int idx = (qc < M) ? qc : (M - 1);
        const float4* fr = (const float4*)(gf + (size_t)idx * 16);
        U16 x0;
#pragma unroll
        for (int v4 = 0; v4 < 4; ++v4) {
          float4 f = fr[v4];
          U2 pA, pB;
          pA.h = __builtin_amdgcn_cvt_pkrtz(f.x, f.y);
          pB.h = __builtin_amdgcn_cvt_pkrtz(f.z, f.w);
          x0.u[v4 * 2 + 0] = pA.u;
          x0.u[v4 * 2 + 1] = pB.u;
        }
        Bx0 = x0.h;
        // chunk1: enc j=16..31 (f=4..7): seed at f=4, 3 doublings
        float sf[2][4], cf[2][4];
#pragma unroll
        for (int d = 0; d < 2; ++d) {
          float p = d ? p1 : p0;
          float s = __sinf(16.0f * PI_F * p), c = __cosf(16.0f * PI_F * p);
          sf[d][0] = s; cf[d][0] = c;
#pragma unroll
          for (int k2 = 1; k2 < 4; ++k2) { dbl_angle(s, c); sf[d][k2] = s; cf[d][k2] = c; }
        }
        U16 x1;
#pragma unroll
        for (int w = 0; w < 8; ++w) {
          int i0 = 2 * w, i1 = 2 * w + 1;
          float vA = (i0 & 1) ? cf[(i0 >> 1) & 1][i0 >> 2] : sf[(i0 >> 1) & 1][i0 >> 2];
          float vB = (i1 & 1) ? cf[(i1 >> 1) & 1][i1 >> 2] : sf[(i1 >> 1) & 1][i1 >> 2];
          U2 p2; p2.h = __builtin_amdgcn_cvt_pkrtz(vA, vB);
          x1.u[w] = p2.u;
        }
        Bx1 = x1.h;
      } else {
        // chunk0: enc j=0..15 (f=0..3); chunk1: enc j=32..39 (f=8..9), 1, zeros
        float sf[2][4], cf[2][4], s8[2][2], c8[2][2];
#pragma unroll
        for (int d = 0; d < 2; ++d) {
          float p = d ? p1 : p0;
          float s = __sinf(PI_F * p), c = __cosf(PI_F * p);
          sf[d][0] = s; cf[d][0] = c;
#pragma unroll
          for (int k2 = 1; k2 < 4; ++k2) { dbl_angle(s, c); sf[d][k2] = s; cf[d][k2] = c; }
          float s2 = __sinf(256.0f * PI_F * p), c2 = __cosf(256.0f * PI_F * p);
          s8[d][0] = s2; c8[d][0] = c2;
          dbl_angle(s2, c2); s8[d][1] = s2; c8[d][1] = c2;
        }
        U16 x0;
#pragma unroll
        for (int w = 0; w < 8; ++w) {
          int i0 = 2 * w, i1 = 2 * w + 1;
          float vA = (i0 & 1) ? cf[(i0 >> 1) & 1][i0 >> 2] : sf[(i0 >> 1) & 1][i0 >> 2];
          float vB = (i1 & 1) ? cf[(i1 >> 1) & 1][i1 >> 2] : sf[(i1 >> 1) & 1][i1 >> 2];
          U2 p2; p2.h = __builtin_amdgcn_cvt_pkrtz(vA, vB);
          x0.u[w] = p2.u;
        }
        Bx0 = x0.h;
        U16 x1;
#pragma unroll
        for (int w = 0; w < 4; ++w) {   // halves 0..7: j=32+i
          int i0 = 2 * w, i1 = 2 * w + 1;
          float vA = (i0 & 1) ? c8[(i0 >> 1) & 1][i0 >> 2] : s8[(i0 >> 1) & 1][i0 >> 2];
          float vB = (i1 & 1) ? c8[(i1 >> 1) & 1][i1 >> 2] : s8[(i1 >> 1) & 1][i1 >> 2];
          U2 p2; p2.h = __builtin_amdgcn_cvt_pkrtz(vA, vB);
          x1.u[w] = p2.u;
        }
        {
          U2 p2; p2.h = __builtin_amdgcn_cvt_pkrtz(1.0f, 0.0f);  // half 8 = 1.0
          x1.u[4] = p2.u;
          x1.u[5] = 0u; x1.u[6] = 0u; x1.u[7] = 0u;
        }
        Bx1 = x1.h;
      }
    }

    // ---- fused layer1 -> layer2, streamed per layer-2 k-chunk c ----
    v8f acc2[4];
#pragma unroll
    for (int c = 0; c < 4; ++c) {
      v8h pA, pB;
      {
        int t = 2 * c;
        v16h A0 = *(const v16h*)&sW[F1_OFF + (t * 2 + 0) * 512 + lane * 16];
        v16h A1 = *(const v16h*)&sW[F1_OFF + (t * 2 + 1) * 512 + lane * 16];
        v8f acc = {};
        acc = wmma_f16(A0, Bx0, acc);
        acc = wmma_f16(A1, Bx1, acc);
        pA = relu_pack(acc);
      }
      {
        int t = 2 * c + 1;
        v16h A0 = *(const v16h*)&sW[F1_OFF + (t * 2 + 0) * 512 + lane * 16];
        v16h A1 = *(const v16h*)&sW[F1_OFF + (t * 2 + 1) * 512 + lane * 16];
        v8f acc = {};
        acc = wmma_f16(A0, Bx0, acc);
        acc = wmma_f16(A1, Bx1, acc);
        pB = relu_pack(acc);
      }
      v16h f2 = shuffle_frag(pA, pB, g);
#pragma unroll
      for (int t2 = 0; t2 < 4; ++t2) {
        v16h A = *(const v16h*)&sW[F2_OFF + (t2 * 4 + c) * 512 + lane * 16];
        if (c == 0) acc2[t2] = wmma_f16(A, f2, bias2v[t2]);
        else        acc2[t2] = wmma_f16(A, f2, acc2[t2]);
      }
    }

    // ---- layer 3: 2 tiles x 2 chunks ----
    v8h packed2[4];
#pragma unroll
    for (int t = 0; t < 4; ++t) packed2[t] = relu_pack(acc2[t]);
    v8f acc3[2];
#pragma unroll
    for (int c = 0; c < 2; ++c) {
      v16h f3 = shuffle_frag(packed2[2 * c], packed2[2 * c + 1], g);
#pragma unroll
      for (int t = 0; t < 2; ++t) {
        v16h A = *(const v16h*)&sW[F3_OFF + (t * 2 + c) * 512 + lane * 16];
        if (c == 0) acc3[t] = wmma_f16(A, f3, bias3v[t]);
        else        acc3[t] = wmma_f16(A, f3, acc3[t]);
      }
    }

    // ---- layer 4: dot over 32 neurons + lane-pair reduce + sigmoid ----
    float part = 0.0f;
#pragma unroll
    for (int t = 0; t < 2; ++t)
#pragma unroll
      for (int r = 0; r < 8; ++r)
        part += fmaxf(acc3[t][r], 0.0f) * w4v[t][r];
    float sum = part + xor16f(part) + b4v;
    float o = 1.0f / (1.0f + __expf(-sum));
    if (g == 0 && q < Q) out[q] = o;
  }
}

// ---------------- launch ----------------
extern "C" void kernel_launch(void* const* d_in, const int* in_sizes, int n_in,
                              void* d_out, int out_size, void* d_ws, size_t ws_size,
                              hipStream_t stream) {
  const float* gf = (const float*)d_in[0];
  const float* qp = (const float*)d_in[1];
  const float* W1 = (const float*)d_in[2];
  const float* b1 = (const float*)d_in[3];
  const float* W2 = (const float*)d_in[4];
  const float* b2 = (const float*)d_in[5];
  const float* W3 = (const float*)d_in[6];
  const float* b3 = (const float*)d_in[7];
  const float* W4 = (const float*)d_in[8];
  const float* b4 = (const float*)d_in[9];

  const int M = in_sizes[0] / 16;
  const int Q = in_sizes[1] / 2;
  const int nTiles = (Q + 15) / 16;

  _Float16* frags = (_Float16*)d_ws;   // 18432 halves = 36 KB

  occdec_prep<<<(FRAG_TOTAL + 255) / 256, 256, 0, stream>>>(W1, b1, W2, W3, frags);

  int blocks = (nTiles + 7) / 8;       // 8 waves per 256-thread block
  if (blocks > 2048) blocks = 2048;
  occdec_main<<<blocks, 256, 0, stream>>>(gf, qp, b2, b3, W4, b4,
                                          frags, (float*)d_out, Q, M, nTiles);
}
